// ANGEL_64252710748241
// MI455X (gfx1250) — matrix operand order repaired and checked by reading, not compiled, not run on hardware
//
#include <hip/hip_runtime.h>
#include <hip/hip_bf16.h>

// ---------------------------------------------------------------------------
// Graph-transformer block on MI455X (gfx1250).
//  - All matmuls via v_wmma_f32_16x16x32_bf16 (fp32 accumulate).
//  - B panels / K panels staged into LDS with the Tensor Data Mover
//    (tensor_load_to_lds + s_wait_tensorcnt), descriptor built per ISA 08 §8.
//  - Sequence padded 257 -> 288 (= 18*16 = 9*32) so all WMMA tilings divide.
// ---------------------------------------------------------------------------

typedef __bf16 bf16;
typedef __attribute__((ext_vector_type(16))) __bf16 v16bf;
typedef __attribute__((ext_vector_type(8)))  __bf16 v8bf;
typedef __attribute__((ext_vector_type(8)))  float  v8f;
typedef __attribute__((ext_vector_type(4)))  unsigned int v4u;
typedef __attribute__((ext_vector_type(8)))  int v8i;
typedef __attribute__((ext_vector_type(4)))  int v4i;

#define GG   256
#define NN   256
#define DD   256
#define HH   8
#define DH   32
#define SREAL 257
#define SP   288                  // padded sequence
#define MP   (GG * SP)            // 73728 rows for padded QKV/attn GEMMs
#define MH   (GG * NN + GG)       // 65792 rows for h / FFN / output

__device__ __forceinline__ v16bf cat16(v8bf a, v8bf b) {
    return __builtin_shufflevector(a, b, 0,1,2,3,4,5,6,7,8,9,10,11,12,13,14,15);
}

__device__ __forceinline__ v8f wmma_bf16(v16bf a, v16bf b, v8f c) {
    return __builtin_amdgcn_wmma_f32_16x16x32_bf16(false, a, false, b,
                                                   (short)0, c, false, false);
}

#if __has_builtin(__builtin_amdgcn_tensor_load_to_lds) && \
    __has_builtin(__builtin_amdgcn_s_wait_tensorcnt)
#define USE_TDM 1
#else
#define USE_TDM 0
#endif

#if USE_TDM
// ---------------------------------------------------------------------------
// 2D TDM load: tile_h rows x tile_w bf16 elements, global row stride
// stride_el elements, into contiguous LDS at byte offset lds_off.
// D# per CDNA5 ISA (count=1, type=2, ds=2B).
// ---------------------------------------------------------------------------
__device__ __forceinline__ void tdm_load_2d(unsigned lds_off, const bf16* gsrc,
                                            unsigned tile_w, unsigned tile_h,
                                            unsigned stride_el) {
    unsigned long long ga = (unsigned long long)(size_t)gsrc;
    v4u g0;
    g0.x = 1u;                                        // count=1 (valid), user mode
    g0.y = lds_off;                                   // lds_addr (bytes)
    g0.z = (unsigned)(ga & 0xFFFFFFFFu);              // global_addr[31:0]
    g0.w = (unsigned)((ga >> 32) & 0x01FFFFFFu)       // global_addr[56:32]
         | 0x80000000u;                               // type=2 ("image")
    v8i g1;
    g1[0] = (int)(1u << 16);                          // data_size=1 (2 bytes)
    g1[1] = (int)((tile_w & 0xFFFFu) << 16);          // tensor_dim0[15:0]
    g1[2] = (int)((tile_w >> 16) |                    // tensor_dim0[31:16]
                  ((tile_h & 0xFFFFu) << 16));        // tensor_dim1[15:0]
    g1[3] = (int)((tile_h >> 16) |                    // tensor_dim1[31:16]
                  (tile_w << 16));                    // tile_dim0
    g1[4] = (int)(tile_h & 0xFFFFu);                  // tile_dim1 (tile_dim2=0)
    g1[5] = (int)stride_el;                           // tensor_dim0_stride[31:0]
    g1[6] = 0;                                        // stride[47:32], dim1_stride lo
    g1[7] = 0;
    v4i z4 = {0, 0, 0, 0};
#if defined(__clang_major__) && (__clang_major__ >= 23)
    v8i z8 = {0, 0, 0, 0, 0, 0, 0, 0};
    __builtin_amdgcn_tensor_load_to_lds(g0, g1, z4, z4, z8, 0);
#else
    __builtin_amdgcn_tensor_load_to_lds(g0, g1, z4, z4, 0);
#endif
}
#endif

// ---------------------------------------------------------------------------
// Weight transpose + fp32->bf16:  dst[n*K + k] = (bf16) src[k*N + n]
// ---------------------------------------------------------------------------
__global__ void tconv_k(const float* __restrict__ src, bf16* __restrict__ dst,
                        int K, int Nn) {
    int idx = blockIdx.x * 256 + threadIdx.x;
    if (idx >= K * Nn) return;
    int n = idx / K, k = idx % K;
    dst[idx] = (bf16)src[(size_t)k * Nn + n];
}

// ---------------------------------------------------------------------------
// Build padded bf16 X: (G, 288, 256).
// ---------------------------------------------------------------------------
__global__ void prepx_k(const float* __restrict__ x, const float* __restrict__ cls,
                        bf16* __restrict__ Xb) {
    size_t idx = (size_t)blockIdx.x * 256 + threadIdx.x;
    int d  = (int)(idx & 255);
    int sg = (int)(idx >> 8);
    int g = sg / SP, s = sg % SP;
    float v = 0.0f;
    if (s < NN)       v = x[((size_t)(g * NN + s) << 8) + d];
    else if (s == NN) v = cls[((size_t)g << 8) + d];
    Xb[idx] = (bf16)v;
}

// ---------------------------------------------------------------------------
// bf16 GEMM:  C[M][N] = A[M][K] @ Bt[N][K]^T + bias
//   MODE 0: bf16 out    MODE 1: bf16 out + ReLU    MODE 2: f32 out
// Workgroup = 8 waves, each computing a 32x64 C tile (256x64 per WG).
// B panel (64 x 256k) staged once per K-phase into LDS by the TDM and shared
// by all waves; A operands stream from global. 8 WMMAs per 32-wide K chunk.
// Grid: (N/64, M/256).  M%256==0, N%64==0, K%256==0 required.
// ---------------------------------------------------------------------------
template <int MODE>
__global__ void __launch_bounds__(256)
gemm_bf16_k(const bf16* __restrict__ A, const bf16* __restrict__ Bt,
            const float* __restrict__ bias, void* __restrict__ Cout,
            int M, int Nn, int K) {
    __shared__ bf16 Bs[64 * 256];                 // 32 KB B panel
    const int lane = threadIdx.x & 31;
    const int wave = threadIdx.x >> 5;
    const int lo = lane & 15, hi = lane >> 4;
    const int row0 = blockIdx.y * 256 + wave * 32;
    const int col0 = blockIdx.x * 64;

    const bf16* ap0 = A + (size_t)(row0 + lo) * K + hi * 8;        // rows +0..15
    const bf16* ap1 = A + (size_t)(row0 + 16 + lo) * K + hi * 8;   // rows +16..31

    v8f z = {0, 0, 0, 0, 0, 0, 0, 0};
    v8f acc[2][4] = {{z, z, z, z}, {z, z, z, z}};

    for (int kp = 0; kp < K; kp += 256) {
#if USE_TDM
        if (wave == 0) {
            unsigned bs_off = (unsigned)(size_t)&Bs[0];  // flat addr low 32b == LDS offset
            tdm_load_2d(bs_off, Bt + (size_t)col0 * K + kp, 256u, 64u, (unsigned)K);
            __builtin_amdgcn_s_wait_tensorcnt(0);
        }
#else
        for (int idx = threadIdx.x; idx < 64 * 32; idx += 256) {
            int r = idx >> 5, c8 = (idx & 31) * 8;
            *(v8bf*)(Bs + r * 256 + c8) =
                *(const v8bf*)(Bt + (size_t)(col0 + r) * K + kp + c8);
        }
#endif
        __syncthreads();

        for (int kc = 0; kc < 256; kc += 32) {
            int kg = kp + kc;
            __builtin_prefetch(ap0 + kg + 256, 0, 1);    // global_prefetch_b8
            v16bf a0 = cat16(*(const v8bf*)(ap0 + kg), *(const v8bf*)(ap0 + kg + 16));
            v16bf a1 = cat16(*(const v8bf*)(ap1 + kg), *(const v8bf*)(ap1 + kg + 16));
#pragma unroll
            for (int nt = 0; nt < 4; nt++) {
                const bf16* bp = Bs + (nt * 16 + lo) * 256 + kc + hi * 8;
                v16bf bv = cat16(*(const v8bf*)bp, *(const v8bf*)(bp + 16));
                acc[0][nt] = wmma_bf16(a0, bv, acc[0][nt]);
                acc[1][nt] = wmma_bf16(a1, bv, acc[1][nt]);
            }
        }
        __syncthreads();
    }

#pragma unroll
    for (int mt = 0; mt < 2; mt++)
#pragma unroll
        for (int nt = 0; nt < 4; nt++) {
            int n = col0 + nt * 16 + lo;
            float bv = bias[n];
#pragma unroll
            for (int r = 0; r < 8; r++) {
                int m = row0 + mt * 16 + r + hi * 8;     // C/D: m = r + 8*(lane/16)
                float v = acc[mt][nt][r] + bv;
                if (MODE == 1) v = fmaxf(v, 0.0f);
                if (MODE == 2) ((float*)Cout)[(size_t)m * Nn + n] = v;
                else           ((bf16*)Cout)[(size_t)m * Nn + n] = (bf16)v;
            }
        }
}

// ---------------------------------------------------------------------------
// Attention: one workgroup per (graph, head). K_h staged by TDM (288x32 tile,
// row stride 256); V_h^T staged manually (transpose). Each wave: 16-row q
// blocks -> 18 score WMMAs, masked softmax (16-lane shfl reductions), 9x2
// A@V WMMAs with probs transposed C-layout -> A-layout via per-wave LDS.
// ---------------------------------------------------------------------------
__global__ void __launch_bounds__(256)
attn_k(const bf16* __restrict__ Q, const bf16* __restrict__ Kg,
       const bf16* __restrict__ Vg, bf16* __restrict__ O) {
    __shared__ bf16 Ks[SP * DH];        // [s][d]   18 KB
    __shared__ bf16 Vt[DH * SP];        // [d][s]   18 KB
    __shared__ bf16 Ps[8 * 16 * 32];    // per-wave prob scratch, 8 KB

    const int g = blockIdx.x >> 3;      // H = 8
    const int h = blockIdx.x & 7;
    const int tid = threadIdx.x;
    const int lane = tid & 31, wave = tid >> 5;
    const int lo = lane & 15, hi = lane >> 4;

#if USE_TDM
    if (wave == 0) {
        unsigned ks_off = (unsigned)(size_t)&Ks[0];
        tdm_load_2d(ks_off, Kg + (size_t)(g * SP) * DD + h * DH,
                    (unsigned)DH, (unsigned)SP, (unsigned)DD);
    }
#endif
    for (int idx = tid; idx < SP * 4; idx += 256) {
        int s = idx >> 2, dc = (idx & 3) * 8;
        size_t goff = (size_t)(g * SP + s) * DD + h * DH + dc;
#if !USE_TDM
        *(v8bf*)(Ks + s * DH + dc) = *(const v8bf*)(Kg + goff);
#endif
        v8bf vv = *(const v8bf*)(Vg + goff);
#pragma unroll
        for (int j = 0; j < 8; j++) Vt[(dc + j) * SP + s] = vv[j];
    }
#if USE_TDM
    if (wave == 0) __builtin_amdgcn_s_wait_tensorcnt(0);
#endif
    __syncthreads();

    bf16* P = Ps + wave * 512;
    const float scale = 0.17677669529663687f;      // 1/sqrt(32)
    v8f z = {0, 0, 0, 0, 0, 0, 0, 0};

    for (int qt = wave; qt < SP / 16; qt += 8) {
        const bf16* qp = Q + (size_t)(g * SP + qt * 16 + lo) * DD + h * DH + hi * 8;
        v16bf qa = cat16(*(const v8bf*)qp, *(const v8bf*)(qp + 16));

        v8f sc[18];
#pragma unroll
        for (int t = 0; t < 18; t++) {
            const bf16* kp = Ks + (t * 16 + lo) * DH + hi * 8;
            v16bf kb = cat16(*(const v8bf*)kp, *(const v8bf*)(kp + 16));
            sc[t] = wmma_bf16(qa, kb, z);
        }

#pragma unroll
        for (int r = 0; r < 8; r++) {
            float mx = -3.0e38f;
#pragma unroll
            for (int t = 0; t < 18; t++) {
                float madd = ((t * 16 + lo) < SREAL) ? 0.0f : -1.0e9f;
                float v = sc[t][r] * scale + madd;
                sc[t][r] = v;
                mx = fmaxf(mx, v);
            }
            mx = fmaxf(mx, __shfl_xor(mx, 1, 32));
            mx = fmaxf(mx, __shfl_xor(mx, 2, 32));
            mx = fmaxf(mx, __shfl_xor(mx, 4, 32));
            mx = fmaxf(mx, __shfl_xor(mx, 8, 32));
            float sum = 0.0f;
#pragma unroll
            for (int t = 0; t < 18; t++) {
                float e = __expf(sc[t][r] - mx);
                sc[t][r] = e;
                sum += e;
            }
            sum += __shfl_xor(sum, 1, 32);
            sum += __shfl_xor(sum, 2, 32);
            sum += __shfl_xor(sum, 4, 32);
            sum += __shfl_xor(sum, 8, 32);
            float inv = 1.0f / sum;
#pragma unroll
            for (int t = 0; t < 18; t++) sc[t][r] *= inv;
        }

        v8f oa[2] = {z, z};
        for (int c = 0; c < 9; c++) {
#pragma unroll
            for (int t2 = 0; t2 < 2; t2++) {
                int tt = 2 * c + t2;
#pragma unroll
                for (int r = 0; r < 8; r++)
                    P[(r + 8 * hi) * 32 + t2 * 16 + lo] = (bf16)sc[tt][r];
            }
            v16bf pa = cat16(*(const v8bf*)(P + lo * 32 + hi * 8),
                             *(const v8bf*)(P + lo * 32 + 16 + hi * 8));
#pragma unroll
            for (int nt = 0; nt < 2; nt++) {
                const bf16* vp = Vt + (nt * 16 + lo) * SP + c * 32 + hi * 8;
                v16bf vb = cat16(*(const v8bf*)vp, *(const v8bf*)(vp + 16));
                oa[nt] = wmma_bf16(pa, vb, oa[nt]);
            }
        }

#pragma unroll
        for (int nt = 0; nt < 2; nt++)
#pragma unroll
            for (int r = 0; r < 8; r++) {
                int mm = qt * 16 + r + 8 * hi;
                O[(size_t)(g * SP + mm) * DD + h * DH + nt * 16 + lo] = (bf16)oa[nt][r];
            }
    }
}

// ---------------------------------------------------------------------------
// LN0: h = LayerNorm(attn_out + x_residual) -> h (f32) + h (bf16).
// ---------------------------------------------------------------------------
__global__ void __launch_bounds__(256)
ln0_k(const float* __restrict__ AO, const float* __restrict__ x,
      const float* __restrict__ cls, const float* __restrict__ gam,
      const float* __restrict__ bet, float* __restrict__ hF,
      bf16* __restrict__ hB) {
    int b = blockIdx.x;
    int g = b / SREAL, s = b % SREAL;
    int d = threadIdx.x;
    float xr = (s < NN) ? x[((size_t)(g * NN + s) << 8) + d]
                        : cls[((size_t)g << 8) + d];
    float v = AO[((size_t)(g * SP + s) << 8) + d] + xr;

    float a = v, q = v * v;
#pragma unroll
    for (int o = 16; o >= 1; o >>= 1) {
        a += __shfl_xor(a, o, 32);
        q += __shfl_xor(q, o, 32);
    }
    __shared__ float s1[8], s2[8];
    int wid = d >> 5, ln = d & 31;
    if (ln == 0) { s1[wid] = a; s2[wid] = q; }
    __syncthreads();
    float ta = 0.0f, tq = 0.0f;
#pragma unroll
    for (int i = 0; i < 8; i++) { ta += s1[i]; tq += s2[i]; }
    float mu  = ta * (1.0f / 256.0f);
    float var = tq * (1.0f / 256.0f) - mu * mu;
    float y = (v - mu) * rsqrtf(var + 1e-5f) * gam[d] + bet[d];

    size_t hrow = (s < NN) ? (size_t)(g * NN + s) : (size_t)(GG * NN + g);
    hF[(hrow << 8) + d] = y;
    hB[(hrow << 8) + d] = (bf16)y;
}

// ---------------------------------------------------------------------------
// LN1: out = LayerNorm(f2 + h).
// ---------------------------------------------------------------------------
__global__ void __launch_bounds__(256)
ln1_k(const float* __restrict__ f2, const float* __restrict__ hF,
      const float* __restrict__ gam, const float* __restrict__ bet,
      float* __restrict__ out) {
    size_t row = blockIdx.x;
    int d = threadIdx.x;
    float v = f2[(row << 8) + d] + hF[(row << 8) + d];
    float a = v, q = v * v;
#pragma unroll
    for (int o = 16; o >= 1; o >>= 1) {
        a += __shfl_xor(a, o, 32);
        q += __shfl_xor(q, o, 32);
    }
    __shared__ float s1[8], s2[8];
    int wid = d >> 5, ln = d & 31;
    if (ln == 0) { s1[wid] = a; s2[wid] = q; }
    __syncthreads();
    float ta = 0.0f, tq = 0.0f;
#pragma unroll
    for (int i = 0; i < 8; i++) { ta += s1[i]; tq += s2[i]; }
    float mu  = ta * (1.0f / 256.0f);
    float var = tq * (1.0f / 256.0f) - mu * mu;
    out[(row << 8) + d] = (v - mu) * rsqrtf(var + 1e-5f) * gam[d] + bet[d];
}

// ---------------------------------------------------------------------------
extern "C" void kernel_launch(void* const* d_in, const int* in_sizes, int n_in,
                              void* d_out, int out_size, void* d_ws, size_t ws_size,
                              hipStream_t stream) {
    (void)in_sizes; (void)n_in; (void)out_size; (void)ws_size;
    const float* x   = (const float*)d_in[0];
    /* d_in[1] (batch) unused: structure is repeat(arange(G), N) */
    const float* cls = (const float*)d_in[2];
    const float* Wq  = (const float*)d_in[3];  const float* bq = (const float*)d_in[4];
    const float* Wk  = (const float*)d_in[5];  const float* bk = (const float*)d_in[6];
    const float* Wv  = (const float*)d_in[7];  const float* bv = (const float*)d_in[8];
    const float* Wo  = (const float*)d_in[9];  const float* bo = (const float*)d_in[10];
    const float* g0  = (const float*)d_in[11]; const float* be0 = (const float*)d_in[12];
    const float* W1  = (const float*)d_in[13]; const float* b1 = (const float*)d_in[14];
    const float* W2  = (const float*)d_in[15]; const float* b2 = (const float*)d_in[16];
    const float* g1  = (const float*)d_in[17]; const float* be1 = (const float*)d_in[18];

    char* ws = (char*)d_ws;
    size_t off = 0;
    auto alloc = [&](size_t bytes) -> void* {
        void* p = ws + off;
        off = (off + bytes + 255) & ~(size_t)255;
        return p;
    };

    bf16* Wqt = (bf16*)alloc((size_t)DD * DD * 2);
    bf16* Wkt = (bf16*)alloc((size_t)DD * DD * 2);
    bf16* Wvt = (bf16*)alloc((size_t)DD * DD * 2);
    bf16* Wot = (bf16*)alloc((size_t)DD * DD * 2);
    bf16* W1t = (bf16*)alloc((size_t)DD * 2 * DD * 2);
    bf16* W2t = (bf16*)alloc((size_t)2 * DD * DD * 2);
    bf16* Xb  = (bf16*)alloc((size_t)MP * DD * 2);
    bf16* Qb  = (bf16*)alloc((size_t)MP * DD * 2);
    bf16* Kb  = (bf16*)alloc((size_t)MP * DD * 2);
    bf16* Vb  = (bf16*)alloc((size_t)MP * DD * 2);
    bf16* Ob  = (bf16*)alloc((size_t)MP * DD * 2);
    float* AO = (float*)alloc((size_t)MP * DD * 4);   // reused as f2 later
    float* hF = (float*)alloc((size_t)MH * DD * 4);
    bf16* hB  = (bf16*)alloc((size_t)MH * DD * 2);
    bf16* f1  = (bf16*)alloc((size_t)MH * 2 * DD * 2);
    float* f2 = AO;

    tconv_k<<<(DD * DD + 255) / 256, 256, 0, stream>>>(Wq, Wqt, DD, DD);
    tconv_k<<<(DD * DD + 255) / 256, 256, 0, stream>>>(Wk, Wkt, DD, DD);
    tconv_k<<<(DD * DD + 255) / 256, 256, 0, stream>>>(Wv, Wvt, DD, DD);
    tconv_k<<<(DD * DD + 255) / 256, 256, 0, stream>>>(Wo, Wot, DD, DD);
    tconv_k<<<(DD * 2 * DD + 255) / 256, 256, 0, stream>>>(W1, W1t, DD, 2 * DD);
    tconv_k<<<(2 * DD * DD + 255) / 256, 256, 0, stream>>>(W2, W2t, 2 * DD, DD);

    prepx_k<<<MP, 256, 0, stream>>>(x, cls, Xb);

    dim3 gqkv(DD / 64, MP / 256);
    gemm_bf16_k<0><<<gqkv, 256, 0, stream>>>(Xb, Wqt, bq, Qb, MP, DD, DD);
    gemm_bf16_k<0><<<gqkv, 256, 0, stream>>>(Xb, Wkt, bk, Kb, MP, DD, DD);
    gemm_bf16_k<0><<<gqkv, 256, 0, stream>>>(Xb, Wvt, bv, Vb, MP, DD, DD);

    attn_k<<<GG * HH, 256, 0, stream>>>(Qb, Kb, Vb, Ob);

    gemm_bf16_k<2><<<gqkv, 256, 0, stream>>>(Ob, Wot, bo, AO, MP, DD, DD);

    ln0_k<<<GG * SREAL, 256, 0, stream>>>(AO, x, cls, g0, be0, hF, hB);

    dim3 gf1(2 * DD / 64, MH / 256);
    gemm_bf16_k<1><<<gf1, 256, 0, stream>>>(hB, W1t, b1, f1, MH, 2 * DD, DD);
    dim3 gf2(DD / 64, MH / 256);
    gemm_bf16_k<2><<<gf2, 256, 0, stream>>>(f1, W2t, b2, f2, MH, DD, 2 * DD);

    ln1_k<<<MH, 256, 0, stream>>>(f2, hF, g1, be1, (float*)d_out);
}
